// MultiHeadAttentionBlock_22789096473374
// MI455X (gfx1250) — compile-verified, matrix operand order checked by reading, not run on hardware
//
#include <hip/hip_runtime.h>
#include <hip/hip_bf16.h>
#include <stdint.h>

// ---------------------------------------------------------------------------
// MultiHeadAttentionBlock for MI455X (gfx1250): all GEMMs via
// v_wmma_f32_16x16x32_bf16; projection/output GEMM uses double-buffered
// GLOBAL_LOAD_ASYNC_TO_LDS_B128 staging synchronized with s_wait_asynccnt;
// flash-style fused attention with LDS float atomics for head-mean logits and
// per-wave LDS transpose of score tiles.
// ---------------------------------------------------------------------------

typedef __bf16 bf16;
typedef __attribute__((ext_vector_type(16))) __bf16 v16bf;
typedef __attribute__((ext_vector_type(8)))  float  v8f;

#define D_MODEL 1024
#define NHEAD   16
#define SEQ     2048
#define DK      64

union BfPack { v16bf v; uint32_t u[8]; };

__device__ __forceinline__ uint32_t pack2bf(float a, float b) {
  unsigned short lo = __builtin_bit_cast(unsigned short, (bf16)a);
  unsigned short hi = __builtin_bit_cast(unsigned short, (bf16)b);
  return (uint32_t)lo | ((uint32_t)hi << 16);
}

__device__ __forceinline__ v8f wmma_bf16(v16bf a, v16bf b, v8f c) {
  // 8 args: (neg_a, A, neg_b, B, c_mod, C, reuse_a, reuse_b)
  return __builtin_amdgcn_wmma_f32_16x16x32_bf16(false, a, false, b,
                                                 (short)0, c, false, false);
}

// Async copy of 16 bytes per lane: global -> LDS, tracked by ASYNCcnt.
__device__ __forceinline__ void async_copy_b128(uint32_t lds_addr,
                                                uint64_t glb_addr) {
  asm volatile("global_load_async_to_lds_b128 %0, %1, off"
               :: "v"(lds_addr), "v"(glb_addr) : "memory");
}
// Wait until at most N of this wave's async ops remain outstanding.
#define WAIT_ASYNC_LE3() asm volatile("s_wait_asynccnt 0x3" ::: "memory")
#define WAIT_ASYNC_0()   asm volatile("s_wait_asynccnt 0x0" ::: "memory")

__device__ __forceinline__ uint32_t lds_off32(const void* p) {
  // Generic pointers into LDS carry the LDS byte offset in the low 32 bits.
  return (uint32_t)(uintptr_t)p;
}

// Load a 16x32 bf16 tile in ISA A-layout (lane = M row; K pairs per VGPR,
// K-halves split at lane 16). rowStrideBytes for padded LDS rows.
__device__ __forceinline__ v16bf load_tileA_any(const bf16* __restrict__ base,
                                                int rowStrideBytes, int row0,
                                                int k0) {
  const int lane = threadIdx.x & 31;
  const int row  = row0 + (lane & 15);
  const int kh   = (lane >> 4) << 3;
  const uint32_t* rp = reinterpret_cast<const uint32_t*>(
      reinterpret_cast<const char*>(base) + (size_t)row * rowStrideBytes);
  BfPack p;
#pragma unroll
  for (int j = 0; j < 8; ++j) {
    int k = k0 + ((j & 4) << 2) + kh + ((j & 3) << 1);
    p.u[j] = rp[(size_t)(k >> 1)];
  }
  return p.v;
}

// Load a 32x16 bf16 B tile where the K dimension strides by `ld` elements in
// memory (V matrix: rows = keys, cols = feature). lane = N column.
__device__ __forceinline__ v16bf load_tileB_strided(const bf16* __restrict__ base,
                                                    int ld, int key0, int col0) {
  const int lane = threadIdx.x & 31;
  const int col  = col0 + (lane & 15);
  const int kh   = (lane >> 4) << 3;
  const unsigned short* sp = reinterpret_cast<const unsigned short*>(base);
  BfPack p;
#pragma unroll
  for (int j = 0; j < 8; ++j) {
    int k = key0 + ((j & 4) << 2) + kh + ((j & 3) << 1);
    uint32_t lo = sp[(size_t)k * ld + col];
    uint32_t hi = sp[(size_t)(k + 1) * ld + col];
    p.u[j] = lo | (hi << 16);
  }
  return p.v;
}

// Read a 16x32 f32 tile from LDS scratch, scale, pack to bf16 A-layout.
__device__ __forceinline__ v16bf load_tileA_lds_f32(const float* sc, float scale) {
  const int lane = threadIdx.x & 31;
  const int r    = lane & 15;
  const int kh   = (lane >> 4) << 3;
  BfPack p;
#pragma unroll
  for (int j = 0; j < 8; ++j) {
    int k = ((j & 4) << 2) + kh + ((j & 3) << 1);
    float2 f = *reinterpret_cast<const float2*>(sc + r * 32 + k);
    p.u[j] = pack2bf(f.x * scale, f.y * scale);
  }
  return p.v;
}

// ---------------------------------------------------------------------------
// Elementwise f32 -> bf16
// ---------------------------------------------------------------------------
__global__ void f32_to_bf16_kernel(const float* __restrict__ in,
                                   bf16* __restrict__ out, int n) {
  int i = blockIdx.x * blockDim.x + threadIdx.x;
  if (i < n) out[i] = (bf16)in[i];
}

// ---------------------------------------------------------------------------
// C[M,N] (f32, + optional bf16 mirror) = A[M,K] @ W[N,K]^T, all bf16 inputs.
// Block = 8 waves, wave grid 4(M) x 2(N) of 16x16 WMMA tiles -> 64x32 block
// tile. K is consumed 64 deep per stage; A (64x64) and W (32x64) panels are
// staged to LDS with double-buffered async b128 copies (3 instructions per
// wave per stage -> s_wait_asynccnt 3 releases the previous stage).
// ---------------------------------------------------------------------------
#define LDS_ROW_BF16 72   /* 64 data + 8 pad -> 144B row stride */
#define LDS_ROW_B    (LDS_ROW_BF16 * 2)

__global__ void __launch_bounds__(256)
gemm_bf16_wmma_kernel(const bf16* __restrict__ A, const bf16* __restrict__ W,
                      float* __restrict__ Cf, bf16* __restrict__ Cb,
                      int M, int N, int Kd) {
  __shared__ bf16 ldsA[2][64 * LDS_ROW_BF16];
  __shared__ bf16 ldsW[2][32 * LDS_ROW_BF16];

  const int tid  = threadIdx.x;
  const int w    = tid >> 5;
  const int lane = tid & 31;
  const int row0 = blockIdx.y * 64;
  const int col0 = blockIdx.x * 32;

  // Stage one 64-deep K panel into LDS buffer `buf`.
  // 24 async b128 instructions per block (16 for A, 8 for W) = 3 per wave.
  auto stage = [&](int buf, int k0) {
#pragma unroll
    for (int i = 0; i < 3; ++i) {
      int base = w * 3 + i;               // 0..23; <16 -> A panel, else W
      int seg  = base * 32 + lane;        // 16B segment index
      uint64_t g;
      uint32_t l;
      if (base < 16) {
        int row = seg >> 3, kp = seg & 7; // 64 rows x 8 chunks
        g = (uint64_t)(uintptr_t)(A + (size_t)(row0 + row) * Kd + k0 + kp * 8);
        l = lds_off32(&ldsA[buf][row * LDS_ROW_BF16 + kp * 8]);
      } else {
        int s2 = seg - 512;
        int row = s2 >> 3, kp = s2 & 7;   // 32 rows x 8 chunks
        g = (uint64_t)(uintptr_t)(W + (size_t)(col0 + row) * Kd + k0 + kp * 8);
        l = lds_off32(&ldsW[buf][row * LDS_ROW_BF16 + kp * 8]);
      }
      async_copy_b128(l, g);
    }
  };

  v8f c;
#pragma unroll
  for (int r = 0; r < 8; ++r) c[r] = 0.0f;

  const int rA = (w & 3) * 16;   // wave's row tile inside A panel
  const int rW = (w >> 2) * 16;  // wave's col tile inside W panel
  const int nIter = Kd / 64;

  stage(0, 0);
  for (int it = 0; it < nIter; ++it) {
    if (it + 1 < nIter) {
      stage((it + 1) & 1, (it + 1) * 64);
      WAIT_ASYNC_LE3();   // previous stage (oldest 3 ops) has landed
    } else {
      WAIT_ASYNC_0();
    }
    __syncthreads();
    const bf16* Ab = ldsA[it & 1];
    const bf16* Wb = ldsW[it & 1];
#pragma unroll
    for (int half = 0; half < 2; ++half) {
      v16bf a = load_tileA_any(Ab, LDS_ROW_B, rA, half * 32);
      v16bf b = load_tileA_any(Wb, LDS_ROW_B, rW, half * 32);
      c = wmma_bf16(a, b, c);
    }
    __syncthreads();
  }

  const int row = row0 + rA + 8 * (lane >> 4);
  const int col = col0 + rW + (lane & 15);
#pragma unroll
  for (int r = 0; r < 8; ++r) {
    float v = c[r];
    size_t idx = (size_t)(row + r) * N + col;
    if (Cf) Cf[idx] = v;
    if (Cb) Cb[idx] = (bf16)v;
  }
}

// ---------------------------------------------------------------------------
// Fused attention: one block = 16 query rows x all 16 heads (wave w -> heads
// 2w,2w+1), streaming 32-key tiles over the full sequence.
//   scores  = Q Kt / 8            (per head, WMMA)
//   logits  = mean_h(scores)      (LDS ds_add_f32 reduction, direct store)
//   rowmax  -> sel = max_k > 0.5
//   new_q  += scores @ V          (WMMA, A from LDS transpose of scores)
//   x       = sel ? new_q : Q     (written bf16 for the output projection)
// ---------------------------------------------------------------------------
__global__ void __launch_bounds__(256)
attn_fused_kernel(const bf16* __restrict__ Qb, const bf16* __restrict__ Kb,
                  const bf16* __restrict__ Vb, const float* __restrict__ Qf,
                  float* __restrict__ logits_out, bf16* __restrict__ xb) {
  __shared__ float logitsAcc[16 * 32];
  __shared__ float scratch[8][16 * 32];
  __shared__ float rowmax[16];
  __shared__ float selv[16];

  const int tid  = threadIdx.x;
  const int w    = tid >> 5;
  const int lane = tid & 31;
  const int q0   = blockIdx.x * 16;
  const float kSc = 0.125f;           // 1/sqrt(dk)
  const float kLg = 0.125f / 16.0f;   // 1/sqrt(dk) * mean over heads

  if (tid < 16) rowmax[tid] = -3.0e38f;
  for (int i = tid; i < 512; i += 256) logitsAcc[i] = 0.0f;
  __syncthreads();

  // Persistent Q A-tiles for this wave's two heads (16 q-rows x 64 d).
  v16bf aq[2][2];
#pragma unroll
  for (int hh = 0; hh < 2; ++hh) {
    int h = 2 * w + hh;
#pragma unroll
    for (int cch = 0; cch < 2; ++cch)
      aq[hh][cch] = load_tileA_any(Qb + h * DK, D_MODEL * 2, q0, cch * 32);
  }

  v8f nq[2][4];
#pragma unroll
  for (int hh = 0; hh < 2; ++hh)
#pragma unroll
    for (int t = 0; t < 4; ++t)
#pragma unroll
      for (int r = 0; r < 8; ++r) nq[hh][t][r] = 0.0f;

  const int rowA = 8 * (lane >> 4);
  const int colA = lane & 15;

  for (int kt = 0; kt < SEQ / 32; ++kt) {
    const int key0 = kt * 32;
#pragma unroll
    for (int hh = 0; hh < 2; ++hh) {
      const int h = 2 * w + hh;
      // raw scores tile, 16 q x 32 keys, accumulated over dk=64 (two WMMAs).
      v8f s[2];
#pragma unroll
      for (int nt = 0; nt < 2; ++nt) {
#pragma unroll
        for (int r = 0; r < 8; ++r) s[nt][r] = 0.0f;
        v16bf b0 = load_tileA_any(Kb + h * DK, D_MODEL * 2, key0 + nt * 16, 0);
        v16bf b1 = load_tileA_any(Kb + h * DK, D_MODEL * 2, key0 + nt * 16, 32);
        s[nt] = wmma_bf16(aq[hh][0], b0, s[nt]);
        s[nt] = wmma_bf16(aq[hh][1], b1, s[nt]);
      }
      // head-sum for logits + per-wave scratch for the C->A transpose.
#pragma unroll
      for (int nt = 0; nt < 2; ++nt)
#pragma unroll
        for (int r = 0; r < 8; ++r) {
          int rr = rowA + r;
          int cc = nt * 16 + colA;
          scratch[w][rr * 32 + cc] = s[nt][r];
          atomicAdd(&logitsAcc[rr * 32 + cc], s[nt][r]);
        }
      __syncthreads();
      // scores (scaled) as bf16 A-tile; new_q += scores @ V.
      v16bf as = load_tileA_lds_f32(scratch[w], kSc);
#pragma unroll
      for (int t = 0; t < 4; ++t) {
        v16bf bv = load_tileB_strided(Vb, D_MODEL, key0, h * DK + t * 16);
        nq[hh][t] = wmma_bf16(as, bv, nq[hh][t]);
      }
      __syncthreads();
    }
    // logits tile out + running row-max, then reset accumulator.
    for (int i = tid; i < 512; i += 256) {
      int rr = i >> 5, cc = i & 31;
      logits_out[(size_t)(q0 + rr) * SEQ + key0 + cc] = logitsAcc[i] * kLg;
    }
    if (tid < 16) {
      float m = rowmax[tid];
#pragma unroll 4
      for (int cc = 0; cc < 32; ++cc) m = fmaxf(m, logitsAcc[tid * 32 + cc] * kLg);
      rowmax[tid] = m;
    }
    __syncthreads();
    for (int i = tid; i < 512; i += 256) logitsAcc[i] = 0.0f;
    __syncthreads();
  }

  if (tid < 16) selv[tid] = (rowmax[tid] > 0.5f) ? 1.0f : 0.0f;
  __syncthreads();

  // x = sel ? new_q : projected Q, stored bf16 for the output projection.
#pragma unroll
  for (int hh = 0; hh < 2; ++hh) {
    const int h = 2 * w + hh;
#pragma unroll
    for (int t = 0; t < 4; ++t)
#pragma unroll
      for (int r = 0; r < 8; ++r) {
        int rl  = rowA + r;
        int row = q0 + rl;
        int d   = t * 16 + colA;
        float qv = Qf[(size_t)row * D_MODEL + h * DK + d];
        float v  = (selv[rl] != 0.0f) ? nq[hh][t][r] : qv;
        xb[(size_t)row * D_MODEL + h * DK + d] = (bf16)v;
      }
  }
}

// ---------------------------------------------------------------------------
// Output permutes: Q -> [h, S, dk], K -> [h, dk, S]
// ---------------------------------------------------------------------------
__global__ void permute_q_kernel(const float* __restrict__ Qf,
                                 float* __restrict__ out3) {
  int i = blockIdx.x * blockDim.x + threadIdx.x;
  int h = i / (SEQ * DK);
  int rem = i % (SEQ * DK);
  int s = rem / DK;
  int d = rem % DK;
  out3[i] = Qf[(size_t)s * D_MODEL + h * DK + d];
}

__global__ void permute_kt_kernel(const float* __restrict__ Kf,
                                  float* __restrict__ out4) {
  int i = blockIdx.x * blockDim.x + threadIdx.x;
  int h = i / (DK * SEQ);
  int rem = i % (DK * SEQ);
  int d = rem / SEQ;
  int s = rem % SEQ;
  out4[i] = Kf[(size_t)s * D_MODEL + h * DK + d];
}

// ---------------------------------------------------------------------------
extern "C" void kernel_launch(void* const* d_in, const int* in_sizes, int n_in,
                              void* d_out, int out_size, void* d_ws, size_t ws_size,
                              hipStream_t stream) {
  (void)in_sizes; (void)n_in; (void)out_size; (void)ws_size;

  const float* q  = (const float*)d_in[0];
  const float* k  = (const float*)d_in[1];
  const float* v  = (const float*)d_in[2];
  /* d_in[3] = mask, unused by the module */
  const float* Wq = (const float*)d_in[4];
  const float* Wk = (const float*)d_in[5];
  const float* Wv = (const float*)d_in[6];
  const float* Wo = (const float*)d_in[7];

  const size_t SD = (size_t)SEQ * D_MODEL;      // 2,097,152
  const size_t DD = (size_t)D_MODEL * D_MODEL;  // 1,048,576

  // workspace carve-up
  char* ws = (char*)d_ws;
  size_t off = 0;
  auto carve = [&](size_t bytes) { char* p = ws + off; off += bytes; return p; };
  bf16*  qb   = (bf16*)carve(SD * sizeof(bf16));
  bf16*  kb   = (bf16*)carve(SD * sizeof(bf16));
  bf16*  vb   = (bf16*)carve(SD * sizeof(bf16));
  bf16*  Wqb  = (bf16*)carve(DD * sizeof(bf16));
  bf16*  Wkb  = (bf16*)carve(DD * sizeof(bf16));
  bf16*  Wvb  = (bf16*)carve(DD * sizeof(bf16));
  bf16*  Wob  = (bf16*)carve(DD * sizeof(bf16));
  float* Qf   = (float*)carve(SD * sizeof(float));
  float* Kf   = (float*)carve(SD * sizeof(float));
  bf16*  Qb   = (bf16*)carve(SD * sizeof(bf16));
  bf16*  Kb   = (bf16*)carve(SD * sizeof(bf16));
  bf16*  Vb   = (bf16*)carve(SD * sizeof(bf16));
  bf16*  xb   = (bf16*)carve(SD * sizeof(bf16));

  float* out_proj = (float*)d_out;              // [S, D]
  float* logits   = out_proj + SD;              // [S, S]
  float* out_Q    = logits + (size_t)SEQ * SEQ; // [h, S, dk]
  float* out_Kt   = out_Q + SD;                 // [h, dk, S]

  // 1) downconvert activations + weights to bf16
  f32_to_bf16_kernel<<<(int)(SD / 256), 256, 0, stream>>>(q,  qb,  (int)SD);
  f32_to_bf16_kernel<<<(int)(SD / 256), 256, 0, stream>>>(k,  kb,  (int)SD);
  f32_to_bf16_kernel<<<(int)(SD / 256), 256, 0, stream>>>(v,  vb,  (int)SD);
  f32_to_bf16_kernel<<<(int)(DD / 256), 256, 0, stream>>>(Wq, Wqb, (int)DD);
  f32_to_bf16_kernel<<<(int)(DD / 256), 256, 0, stream>>>(Wk, Wkb, (int)DD);
  f32_to_bf16_kernel<<<(int)(DD / 256), 256, 0, stream>>>(Wv, Wvb, (int)DD);
  f32_to_bf16_kernel<<<(int)(DD / 256), 256, 0, stream>>>(Wo, Wob, (int)DD);

  // 2) projections: X @ W^T  (grid: N/32 x M/64)
  dim3 ggrid(D_MODEL / 32, SEQ / 64);
  gemm_bf16_wmma_kernel<<<ggrid, 256, 0, stream>>>(qb, Wqb, Qf, Qb, SEQ, D_MODEL, D_MODEL);
  gemm_bf16_wmma_kernel<<<ggrid, 256, 0, stream>>>(kb, Wkb, Kf, Kb, SEQ, D_MODEL, D_MODEL);
  gemm_bf16_wmma_kernel<<<ggrid, 256, 0, stream>>>(vb, Wvb, (float*)nullptr, Vb, SEQ, D_MODEL, D_MODEL);

  // 3) fused attention (128 blocks of 16 query rows; all heads per block)
  attn_fused_kernel<<<SEQ / 16, 256, 0, stream>>>(Qb, Kb, Vb, Qf, logits, xb);

  // 4) output projection: out = x @ Wo^T
  gemm_bf16_wmma_kernel<<<ggrid, 256, 0, stream>>>(xb, Wob, out_proj, (bf16*)nullptr, SEQ, D_MODEL, D_MODEL);

  // 5) Q and K^T output views
  permute_q_kernel<<<(int)(SD / 256), 256, 0, stream>>>(Qf, out_Q);
  permute_kt_kernel<<<(int)(SD / 256), 256, 0, stream>>>(Kf, out_Kt);
}
